// DeformableConv2D_18399639896410
// MI455X (gfx1250) — matrix operand order repaired and checked by reading, not compiled, not run on hardware
//
#include <hip/hip_runtime.h>

// Problem constants (match reference setup_inputs)
#define GNUM 4
#define BATCH 4
#define HDIM 96
#define WDIM 96
#define CDIM 256
#define CG 64          // CDIM / GNUM
#define K2C 576        // 3*3*CG
#define GF 64          // output channels per group
#define NPIX (BATCH*HDIM*WDIM)   // 36864

#define AS1 __attribute__((address_space(1)))
#define AS3 __attribute__((address_space(3)))

typedef __attribute__((ext_vector_type(16))) __bf16 v16bf;
typedef __attribute__((ext_vector_type(8)))  __bf16 v8bf;
typedef __attribute__((ext_vector_type(8)))  float  v8f;
typedef int v4i __attribute__((__vector_size__(16)));   // matches async-LDS builtin param

__device__ __forceinline__ __bf16 f2bf(float f) {
  union { float f; unsigned u; } v; v.f = f;
  unsigned r = v.u + 0x7FFFu + ((v.u >> 16) & 1u);   // round-to-nearest-even
  union { unsigned short s; __bf16 b; } o; o.s = (unsigned short)(r >> 16);
  return o.b;
}

// ---------------------------------------------------------------------------
// bf16 GEMM with bias: Cout[m, col_off+n] = sum_k A[m,k]*B[k,n] + bias[n]
// Abf: [M x Kd] bf16 row-major.  Btb: [N x Kd] bf16 (B pre-transposed).
// One wave32 computes a 32x32 tile = 2x2 WMMA tiles; 4 v_wmma per k-step.
// Fragment layouts used here (wave32; A/B contiguous k, C/D per CDNA5 ISA 7.12.2):
//   A elem e -> K = half*16 + e  (two 16B contiguous chunks)
//   B lane l15 = col, elems e -> K = k0 + half*16 + e (one 32B chunk in Btb)
//   C/D VGPR i -> row = tile + half*8 + i, col = l15
// ---------------------------------------------------------------------------
__global__ void wmma_gemm_bf16_kernel(const __bf16* __restrict__ Abf,
                                      const __bf16* __restrict__ Btb,
                                      const float* __restrict__ bias,
                                      float* __restrict__ Cout,
                                      int M, int N, int Kd, int ldc, int col_off)
{
  const int lane = threadIdx.x & 31;
  const int wave = (blockIdx.x * blockDim.x + threadIdx.x) >> 5;
  const int tilesN = N >> 5;
  const int totalTiles = (M >> 5) * tilesN;
  if (wave >= totalTiles) return;
  const int tM = (wave / tilesN) << 5;
  const int tN = (wave % tilesN) << 5;
  const int half = lane >> 4;
  const int l15  = lane & 15;

  v8f acc00 = {}, acc01 = {}, acc10 = {}, acc11 = {};
  const __bf16* a0p = Abf + (size_t)(tM + l15) * Kd + half * 16;
  const __bf16* a1p = a0p + (size_t)16 * Kd;
  const __bf16* b0p = Btb + (size_t)(tN + l15) * Kd + half * 16;
  const __bf16* b1p = b0p + (size_t)16 * Kd;

  for (int k0 = 0; k0 < Kd; k0 += 32) {
    if (k0 + 32 < Kd) {
      __builtin_prefetch(a0p + k0 + 32, 0, 1);   // global_prefetch_b8
      __builtin_prefetch(a1p + k0 + 32, 0, 1);
    }
    v8bf a0lo = *(const v8bf*)(a0p + k0);
    v8bf a0hi = *(const v8bf*)(a0p + k0 + 8);
    v8bf a1lo = *(const v8bf*)(a1p + k0);
    v8bf a1hi = *(const v8bf*)(a1p + k0 + 8);
    v16bf a0, a1;
    #pragma unroll
    for (int e = 0; e < 8; ++e) {
      a0[e] = a0lo[e]; a0[e + 8] = a0hi[e];
      a1[e] = a1lo[e]; a1[e + 8] = a1hi[e];
    }
    v16bf b0 = *(const v16bf*)(b0p + k0);
    v16bf b1 = *(const v16bf*)(b1p + k0);

    acc00 = __builtin_amdgcn_wmma_f32_16x16x32_bf16(false, a0, false, b0, (short)0, acc00, false, false);
    acc01 = __builtin_amdgcn_wmma_f32_16x16x32_bf16(false, a0, false, b1, (short)0, acc01, false, false);
    acc10 = __builtin_amdgcn_wmma_f32_16x16x32_bf16(false, a1, false, b0, (short)0, acc10, false, false);
    acc11 = __builtin_amdgcn_wmma_f32_16x16x32_bf16(false, a1, false, b1, (short)0, acc11, false, false);
  }

  const float bv0 = bias[tN + l15];
  const float bv1 = bias[tN + 16 + l15];
  float* c00 = Cout + (size_t)(tM + half * 8) * ldc + col_off + tN + l15;
  float* c10 = c00 + (size_t)16 * ldc;
  #pragma unroll
  for (int i = 0; i < 8; ++i) {
    c00[(size_t)i * ldc]      = acc00[i] + bv0;
    c00[(size_t)i * ldc + 16] = acc01[i] + bv1;
    c10[(size_t)i * ldc]      = acc10[i] + bv0;
    c10[(size_t)i * ldc + 16] = acc11[i] + bv1;
  }
}

// ---------------------------------------------------------------------------
// Elementwise f32 -> bf16 (vectorized by 4)
// ---------------------------------------------------------------------------
__global__ void f32_to_bf16_kernel(const float* __restrict__ in,
                                   __bf16* __restrict__ out, long n)
{
  long base = ((long)blockIdx.x * blockDim.x + threadIdx.x) * 4;
  if (base + 3 < n) {
    float4 v = *(const float4*)(in + base);
    out[base]     = f2bf(v.x); out[base + 1] = f2bf(v.y);
    out[base + 2] = f2bf(v.z); out[base + 3] = f2bf(v.w);
  } else {
    for (long j = base; j < n; ++j) out[j] = f2bf(in[j]);
  }
}

// ---------------------------------------------------------------------------
// Transpose-convert: in [K x N] f32 -> out [N x K] bf16
// ---------------------------------------------------------------------------
__global__ void transpose_bf16_kernel(const float* __restrict__ in,
                                      __bf16* __restrict__ out, int K, int N)
{
  int i = blockIdx.x * blockDim.x + threadIdx.x;
  if (i >= K * N) return;
  int k = i / N, n = i % N;
  out[(size_t)n * K + k] = f2bf(in[i]);
}

// ---------------------------------------------------------------------------
// Offset conv: 3x3 SAME, Cg=64 -> 18 channels, per group.
// Weights staged in LDS via gfx1250 async global->LDS copy when available
// (GLOBAL_LOAD_ASYNC_TO_LDS_B128, tracked by ASYNCcnt).
// ---------------------------------------------------------------------------
__global__ void offset_conv_kernel(const float* __restrict__ y0,
                                   const float* __restrict__ w_off,
                                   const float* __restrict__ b_off,
                                   float* __restrict__ offs)
{
  __shared__ float sW[3*3*CG*18];   // 10368 floats = 41.5 KB
  const int g   = blockIdx.y;
  const int tid = threadIdx.x;
  const float* wg = w_off + (size_t)g * (3*3*CG*18);
#if __has_builtin(__builtin_amdgcn_global_load_async_to_lds_b128)
  {
    AS1 v4i* gsrc = (AS1 v4i*)(v4i*)wg;
    AS3 v4i* ldst = (AS3 v4i*)(v4i*)sW;
    const int nvec = 3*3*CG*18*4/16;   // 2592 x 16B
    for (int i = tid; i < nvec; i += blockDim.x)
      __builtin_amdgcn_global_load_async_to_lds_b128(gsrc + i, ldst + i, 0, 0);
#if __has_builtin(__builtin_amdgcn_s_wait_asynccnt)
    __builtin_amdgcn_s_wait_asynccnt(0);
#else
    asm volatile("s_wait_asynccnt 0x0" ::: "memory");
#endif
  }
#else
  for (int i = tid; i < 3*3*CG*18; i += blockDim.x) sW[i] = wg[i];
#endif
  __syncthreads();

  const int p = blockIdx.x * blockDim.x + tid;
  if (p >= NPIX) return;
  const int b   = p / (HDIM*WDIM);
  const int rem = p % (HDIM*WDIM);
  const int h   = rem / WDIM;
  const int w   = rem % WDIM;

  float acc[18];
  #pragma unroll
  for (int i = 0; i < 18; ++i) acc[i] = b_off[g*18 + i];

  for (int ky = 0; ky < 3; ++ky) {
    int yy = h + ky - 1; if (yy < 0 || yy >= HDIM) continue;
    for (int kx = 0; kx < 3; ++kx) {
      int xx = w + kx - 1; if (xx < 0 || xx >= WDIM) continue;
      const float* xp = y0 + ((size_t)(b*HDIM + yy)*WDIM + xx) * CDIM + g*CG;
      const float* wp = sW + (ky*3 + kx) * CG * 18;
      for (int c = 0; c < CG; ++c) {
        float xv = xp[c];
        const float* wr = wp + c*18;
        #pragma unroll
        for (int oc = 0; oc < 18; ++oc) acc[oc] += xv * wr[oc];
      }
    }
  }
  float* op = offs + ((size_t)g * NPIX + p) * 18;
  #pragma unroll
  for (int oc = 0; oc < 18; ++oc) op[oc] = acc[oc];
}

// ---------------------------------------------------------------------------
// Deformable bilinear sampling, one thread per (pixel, tap). Exact replication
// of the reference clip/floor/bilinear formula. Gathers 4 corners x 64 ch.
// ---------------------------------------------------------------------------
__global__ void sample_kernel(const float* __restrict__ xin,   // y0, [NPIX x CDIM]
                              const float* __restrict__ offs,
                              float* __restrict__ smp,         // [NPIX x K2C]
                              int g)
{
  int t = blockIdx.x * blockDim.x + threadIdx.x;
  if (t >= NPIX * 9) return;
  const int p   = t % 9;
  const int pix = t / 9;
  const int b   = pix / (HDIM*WDIM);
  const int rem = pix % (HDIM*WDIM);
  const int h   = rem / WDIM;
  const int w   = rem % WDIM;
  const int iy  = p / 3, ix = p % 3;

  const float* op = offs + ((size_t)g * NPIX + pix) * 18 + 2*p;
  float lx = (float)w + (float)(ix - 1) + op[0];
  float ly = (float)h + (float)(iy - 1) + op[1];
  lx = fminf(fmaxf(lx, 0.0f), (float)(WDIM - 1));
  ly = fminf(fmaxf(ly, 0.0f), (float)(HDIM - 1));
  float x0f = fminf(fmaxf(floorf(lx), 0.0f), (float)(WDIM - 1));
  float y0f = fminf(fmaxf(floorf(ly), 0.0f), (float)(HDIM - 1));
  float x1f = fminf(x0f + 1.0f, (float)(WDIM - 1));
  float y1f = fminf(y0f + 1.0f, (float)(HDIM - 1));
  float wa = (x1f - lx) * (y1f - ly);
  float wb = (x1f - lx) * (ly - y0f);
  float wc = (lx - x0f) * (y1f - ly);
  float wd = (lx - x0f) * (ly - y0f);
  int x0 = (int)x0f, y0i = (int)y0f, x1 = (int)x1f, y1i = (int)y1f;

  const float* base = xin + (size_t)g * CG;
  const float* pa  = base + ((size_t)(b*HDIM + y0i)*WDIM + x0) * CDIM;
  const float* pbp = base + ((size_t)(b*HDIM + y1i)*WDIM + x0) * CDIM;
  const float* pc  = base + ((size_t)(b*HDIM + y0i)*WDIM + x1) * CDIM;
  const float* pd  = base + ((size_t)(b*HDIM + y1i)*WDIM + x1) * CDIM;
  float* out = smp + (size_t)pix * K2C + p * CG;
  #pragma unroll 4
  for (int c = 0; c < CG; c += 4) {
    float4 va = *(const float4*)(pa + c);
    float4 vb = *(const float4*)(pbp + c);
    float4 vc = *(const float4*)(pc + c);
    float4 vd = *(const float4*)(pd + c);
    float4 r;
    r.x = wa*va.x + wb*vb.x + wc*vc.x + wd*vd.x;
    r.y = wa*va.y + wb*vb.y + wc*vc.y + wd*vd.y;
    r.z = wa*va.z + wb*vb.z + wc*vc.z + wd*vd.z;
    r.w = wa*va.w + wb*vb.w + wc*vc.w + wd*vd.w;
    *(float4*)(out + c) = r;
  }
}

// ---------------------------------------------------------------------------
// Depthwise 3x3 SAME over K2C=576 channels; one thread = (pixel, 4 channels).
// Emits bf16 so the following pointwise WMMA GEMM reads A directly.
// ---------------------------------------------------------------------------
__global__ void dw_kernel(const float* __restrict__ smp,
                          const float* __restrict__ w_dw,
                          const float* __restrict__ b_dw,
                          __bf16* __restrict__ hbf, int g)
{
  const int NC4 = K2C / 4;   // 144
  int t = blockIdx.x * blockDim.x + threadIdx.x;
  if (t >= NPIX * NC4) return;
  int c4  = t % NC4;
  int pix = t / NC4;
  int b   = pix / (HDIM*WDIM);
  int rem = pix % (HDIM*WDIM);
  int h = rem / WDIM, w = rem % WDIM;
  int c = c4 * 4;
  const float* wg = w_dw + (size_t)g * (9*K2C);
  const float* bg = b_dw + (size_t)g * K2C;

  float4 acc;
  acc.x = bg[c]; acc.y = bg[c+1]; acc.z = bg[c+2]; acc.w = bg[c+3];
  for (int ky = 0; ky < 3; ++ky) {
    int yy = h + ky - 1; if (yy < 0 || yy >= HDIM) continue;
    for (int kx = 0; kx < 3; ++kx) {
      int xx = w + kx - 1; if (xx < 0 || xx >= WDIM) continue;
      float4 s  = *(const float4*)(smp + ((size_t)(b*HDIM + yy)*WDIM + xx) * K2C + c);
      float4 wt = *(const float4*)(wg + (size_t)(ky*3 + kx) * K2C + c);
      acc.x += s.x*wt.x; acc.y += s.y*wt.y; acc.z += s.z*wt.z; acc.w += s.w*wt.w;
    }
  }
  __bf16* outp = hbf + (size_t)pix * K2C + c;
  outp[0] = f2bf(acc.x); outp[1] = f2bf(acc.y);
  outp[2] = f2bf(acc.z); outp[3] = f2bf(acc.w);
}

// ---------------------------------------------------------------------------
static inline size_t align256(size_t x) { return (x + 255) & ~(size_t)255; }

extern "C" void kernel_launch(void* const* d_in, const int* in_sizes, int n_in,
                              void* d_out, int out_size, void* d_ws, size_t ws_size,
                              hipStream_t stream) {
  const float* x      = (const float*)d_in[0];
  const float* w_init = (const float*)d_in[1];
  const float* b_init = (const float*)d_in[2];
  const float* w_off  = (const float*)d_in[3];
  const float* b_off  = (const float*)d_in[4];
  const float* w_dw   = (const float*)d_in[5];
  const float* b_dw   = (const float*)d_in[6];
  const float* w_pw   = (const float*)d_in[7];
  const float* b_pw   = (const float*)d_in[8];
  float* out = (float*)d_out;

  // Workspace carve-up (256B aligned regions)
  char* base = (char*)d_ws;
  size_t off = 0;
  float*  y0    = (float*) (base + off); off = align256(off + (size_t)NPIX*CDIM*4);
  float*  offs  = (float*) (base + off); off = align256(off + (size_t)GNUM*NPIX*18*4);
  float*  smp   = (float*) (base + off); off = align256(off + (size_t)NPIX*K2C*4);
  __bf16* xbf   = (__bf16*)(base + off); off = align256(off + (size_t)NPIX*CDIM*2);
  __bf16* hbf   = (__bf16*)(base + off); off = align256(off + (size_t)NPIX*K2C*2);
  __bf16* witb  = (__bf16*)(base + off); off = align256(off + (size_t)CDIM*CDIM*2);
  __bf16* wpwtb = (__bf16*)(base + off); off = align256(off + (size_t)GNUM*GF*K2C*2);

  // 0) precision prep: x -> bf16, w_init -> bf16 transposed, w_pw -> bf16 transposed
  {
    long n = (long)NPIX * CDIM;
    long thr = (n + 3) / 4;
    f32_to_bf16_kernel<<<(int)((thr + 255) / 256), 256, 0, stream>>>(x, xbf, n);
    transpose_bf16_kernel<<<(CDIM*CDIM + 255) / 256, 256, 0, stream>>>(w_init, witb, CDIM, CDIM);
    for (int g = 0; g < GNUM; ++g)
      transpose_bf16_kernel<<<(K2C*GF + 255) / 256, 256, 0, stream>>>(
          w_pw + (size_t)g * K2C * GF, wpwtb + (size_t)g * GF * K2C, K2C, GF);
  }

  // 1) init 1x1 conv: GEMM [NPIX x 256] x [256 x 256] + bias -> y0 (f32)
  {
    int waves  = (NPIX / 32) * (CDIM / 32);
    int blocks = (waves + 7) / 8;
    wmma_gemm_bf16_kernel<<<blocks, 256, 0, stream>>>(
        xbf, witb, b_init, y0, NPIX, CDIM, CDIM, CDIM, 0);
  }
  // 2) offset convs, all groups
  {
    dim3 grid(NPIX / 256, GNUM);
    offset_conv_kernel<<<grid, 256, 0, stream>>>(y0, w_off, b_off, offs);
  }
  // 3) per-group: sample -> depthwise(bf16) -> pointwise WMMA GEMM into d_out
  for (int g = 0; g < GNUM; ++g) {
    int t1 = NPIX * 9;
    sample_kernel<<<(t1 + 255) / 256, 256, 0, stream>>>(y0, offs, smp, g);
    int t2 = NPIX * (K2C / 4);
    dw_kernel<<<(t2 + 255) / 256, 256, 0, stream>>>(smp, w_dw, b_dw, hbf, g);
    int waves  = (NPIX / 32) * (GF / 32);
    int blocks = (waves + 7) / 8;
    wmma_gemm_bf16_kernel<<<blocks, 256, 0, stream>>>(
        hbf, wpwtb + (size_t)g * GF * K2C, b_pw + (size_t)g * GF,
        out, NPIX, GF, K2C, CDIM, g * GF);
  }
}
